// GroupedMLP_44367012168174
// MI455X (gfx1250) — matrix operand order repaired and checked by reading, not compiled, not run on hardware
//
#include <hip/hip_runtime.h>
#include <hip/hip_bf16.h>

// ---------------------------------------------------------------------------
// Grouped MoE MLP (FC1 + SwiGLU fused, then FC2) for MI455X (gfx1250).
//   E=8 experts, HID=2048, FF=4096, N=16384 tokens (contiguous per expert).
// Strategy: bf16 WMMA (v_wmma_f32_16x16x32_bf16) with f32 accumulation.
//   - f32 -> bf16 conversion done with one v_perm_b32 per packed pair.
//   - FC1 block computes matching gate & up column tiles, applies SwiGLU in
//     registers, stores bf16 activations to workspace (no f32 intermediate).
//   - FC2 stages its (already bf16) A tile with global_load_async_to_lds_b128
//     (ASYNCcnt-tracked direct-to-LDS copy, no VGPR round trip).
//   - Token counts are multiples of the 128-row tile, so a tile never
//     straddles an expert boundary (standard MoE grouped-GEMM alignment).
// ---------------------------------------------------------------------------

#define HID 2048
#define FF  4096
#define NEXP 8
#define NTOK 16384

#define TILE_M 128
#define TILE_N 64
#define TILE_K 32
#define A_STRIDE 18   // uints per LDS A row (16 data + 2 pad)
#define B_STRIDE 20   // uints per LDS B row (16 data + 4 pad)

typedef __attribute__((ext_vector_type(16))) __bf16        v16bf;
typedef __attribute__((ext_vector_type(8)))  float         v8f;
typedef __attribute__((ext_vector_type(8)))  unsigned int  v8u;

// float -> bf16 bits (round-to-nearest, ties-away via +0x8000 bias)
__device__ __forceinline__ unsigned bfbits(float x) {
  union { float f; unsigned u; } v; v.f = x;
  return (v.u + 0x8000u) >> 16;
}
// pack two f32 -> {bf16(b), bf16(a)} with a single v_perm_b32
__device__ __forceinline__ unsigned pack2(float a, float b) {
  union { float f; unsigned u; } va, vb; va.f = a; vb.f = b;
  const unsigned ra = va.u + 0x8000u;
  const unsigned rb = vb.u + 0x8000u;
  // {S0=rb bytes 7..4, S1=ra bytes 3..0}; select rb[31:16] : ra[31:16]
  return __builtin_amdgcn_perm(rb, ra, 0x07060302u);
}

// A-fragment gather, contiguous k order used here (same as frag_b; not the ISA table order):
//   lanes 0-15 : M=lane,    K=0..15  (2 per VGPR)
//   lanes 16-31: M=lane-16, K=16..31
// LDS tile is [row][K] with K packed 2-per-uint, row stride A_STRIDE uints.
__device__ __forceinline__ v16bf frag_a(const unsigned int* s, int rowbase, int lane) {
  const int r  = rowbase + (lane & 15);
  const int hi = lane >> 4;
  const unsigned int* p = s + r * A_STRIDE + hi * 8;
  v8u raw;
#pragma unroll
  for (int i = 0; i < 4; ++i) { raw[i] = p[i]; raw[4 + i] = p[4 + i]; }
  v16bf f; __builtin_memcpy(&f, &raw, 32);
  return f;
}

// B-fragment gather, contiguous k order used here (same as frag_a; not the ISA table order):
//   lanes 0-15 : N=lane,    K=0..15  (2 per VGPR)
//   lanes 16-31: N=lane-16, K=16..31
// LDS tile is stored TRANSPOSED as [n][K], row stride B_STRIDE uints.
__device__ __forceinline__ v16bf frag_b(const unsigned int* s, int colbase, int lane) {
  const int n  = colbase + (lane & 15);
  const int hi = lane >> 4;
  const unsigned int* p = s + n * B_STRIDE + hi * 8;
  v8u raw;
#pragma unroll
  for (int i = 0; i < 8; ++i) raw[i] = p[i];
  v16bf f; __builtin_memcpy(&f, &raw, 32);
  return f;
}

__device__ __forceinline__ v8f wmma_bf16(v16bf a, v16bf b, v8f c) {
  return __builtin_amdgcn_wmma_f32_16x16x32_bf16(
      /*neg_a=*/false, a, /*neg_b=*/false, b,
      /*c_mod=*/(short)0, c, /*reuse_a=*/false, /*reuse_b=*/false);
}

// Uniform per-block expert lookup (tokens contiguous per expert).
__device__ __forceinline__ int expert_of_row(const int* __restrict__ tpe, int row) {
  int e = 0, base = 0;
  for (int i = 0; i < NEXP - 1; ++i) {
    int c = tpe[i];
    if (row >= base + c) { base += c; ++e; } else break;
  }
  return e;
}

// ---------------------------------------------------------------------------
// FC1 + SwiGLU: act[n, c] = silu(x @ W1[:, c]) * (x @ W1[:, FF + c]), bf16 out
// ---------------------------------------------------------------------------
__global__ __launch_bounds__(256, 1)
void moe_fc1_swiglu(const float* __restrict__ x,
                    const int* __restrict__ tpe,
                    const float* __restrict__ w1,
                    unsigned short* __restrict__ act) {
  __shared__ unsigned int sA[TILE_M * A_STRIDE];
  __shared__ unsigned int sG[TILE_N * B_STRIDE];
  __shared__ unsigned int sU[TILE_N * B_STRIDE];

  const int m0 = blockIdx.x * TILE_M;
  const int c0 = blockIdx.y * TILE_N;
  const int e  = expert_of_row(tpe, m0);
  const float* __restrict__ W = w1 + (size_t)e * HID * (2 * FF);

  const int t    = threadIdx.x;
  const int lane = t & 31;
  const int wave = t >> 5;
  const int wm   = wave & 3;   // 4 row groups of 32
  const int wn   = wave >> 2;  // 2 col groups of 32

  v8f accg[2][2], accu[2][2];
#pragma unroll
  for (int i = 0; i < 2; ++i)
#pragma unroll
    for (int j = 0; j < 2; ++j) { accg[i][j] = (v8f)0.f; accu[i][j] = (v8f)0.f; }

  // staging coordinates (fixed per thread)
  const int a_cc = (t & 7) * 4;   // 4-wide f32 chunk in K
  const int a_r0 = t >> 3;        // row 0..31 (x4 iterations)
  const int b_n  = t & 63;        // output column within tile
  const int b_kb = (t >> 6) * 8;  // 8-deep K slab

  for (int k0 = 0; k0 < HID; k0 += TILE_K) {
    // ---- stage A (tokens), f32 -> bf16, [row][K] ----
#pragma unroll
    for (int i = 0; i < 4; ++i) {
      const int r = a_r0 + 32 * i;
      const float4 v = *(const float4*)(x + (size_t)(m0 + r) * HID + k0 + a_cc);
      sA[r * A_STRIDE + (a_cc >> 1) + 0] = pack2(v.x, v.y);
      sA[r * A_STRIDE + (a_cc >> 1) + 1] = pack2(v.z, v.w);
    }
    // ---- stage B gate & up, f32 -> bf16, transposed [n][K] ----
    {
      const float* pg = W + (size_t)(k0 + b_kb) * (2 * FF) + c0 + b_n;
      const float* pu = pg + FF;
      if (k0 + TILE_K < HID) {  // uniform branch: EXEC stays all-ones
        __builtin_prefetch(pg + (size_t)TILE_K * (2 * FF), 0, 1);
        __builtin_prefetch(pu + (size_t)TILE_K * (2 * FF), 0, 1);
      }
      float g[8], u[8];
#pragma unroll
      for (int i = 0; i < 8; ++i) {
        g[i] = pg[(size_t)i * (2 * FF)];
        u[i] = pu[(size_t)i * (2 * FF)];
      }
#pragma unroll
      for (int i = 0; i < 4; ++i) {
        sG[b_n * B_STRIDE + (b_kb >> 1) + i] = pack2(g[2 * i], g[2 * i + 1]);
        sU[b_n * B_STRIDE + (b_kb >> 1) + i] = pack2(u[2 * i], u[2 * i + 1]);
      }
    }
    __syncthreads();

    v16bf af[2], gf[2], uf[2];
#pragma unroll
    for (int i = 0; i < 2; ++i) af[i] = frag_a(sA, wm * 32 + i * 16, lane);
#pragma unroll
    for (int j = 0; j < 2; ++j) {
      gf[j] = frag_b(sG, wn * 32 + j * 16, lane);
      uf[j] = frag_b(sU, wn * 32 + j * 16, lane);
    }
#pragma unroll
    for (int i = 0; i < 2; ++i)
#pragma unroll
      for (int j = 0; j < 2; ++j) {
        accg[i][j] = wmma_bf16(af[i], gf[j], accg[i][j]);
        accu[i][j] = wmma_bf16(af[i], uf[j], accu[i][j]);
      }
    __syncthreads();
  }

  // ---- epilogue: SwiGLU in registers, bf16 store to workspace ----
  // C/D layout: lane 0-15 (N=lane, M=v), lane 16-31 (N=lane-16, M=v+8)
  const int rhi = (lane >> 4) * 8;
  const int nc  = lane & 15;
#pragma unroll
  for (int i = 0; i < 2; ++i)
#pragma unroll
    for (int j = 0; j < 2; ++j) {
      const int gm = m0 + wm * 32 + i * 16 + rhi;
      const int gn = c0 + wn * 32 + j * 16 + nc;
#pragma unroll
      for (int v = 0; v < 8; ++v) {
        const float g = accg[i][j][v];
        const float u = accu[i][j][v];
        const float a = (g / (1.f + __expf(-g))) * u;  // silu(g) * u
        act[(size_t)(gm + v) * FF + gn] = (unsigned short)bfbits(a);
      }
    }
}

// ---------------------------------------------------------------------------
// FC2: out = act(bf16) @ W2[e], f32 out
// ---------------------------------------------------------------------------
__global__ __launch_bounds__(256, 1)
void moe_fc2(const unsigned short* __restrict__ act,
             const int* __restrict__ tpe,
             const float* __restrict__ w2,
             float* __restrict__ out) {
  __shared__ unsigned int sA[TILE_M * A_STRIDE];
  __shared__ unsigned int sB[TILE_N * B_STRIDE];

  const int m0 = blockIdx.x * TILE_M;
  const int c0 = blockIdx.y * TILE_N;
  const int e  = expert_of_row(tpe, m0);
  const float* __restrict__ W = w2 + (size_t)e * FF * HID;

  const int t    = threadIdx.x;
  const int lane = t & 31;
  const int wave = t >> 5;
  const int wm   = wave & 3;
  const int wn   = wave >> 2;

  v8f acc[2][2];
#pragma unroll
  for (int i = 0; i < 2; ++i)
#pragma unroll
    for (int j = 0; j < 2; ++j) acc[i][j] = (v8f)0.f;

  const int a_r  = t >> 1;         // row 0..127
  const int a_q0 = (t & 1) * 2;    // uint4 quarters {0,1} or {2,3}
  const int b_n  = t & 63;
  const int b_kb = (t >> 6) * 8;

  // LDS byte offsets for this thread's async-copy destinations (wave-relative:
  // low 32 bits of a generic pointer to LDS are the LDS byte address).
  unsigned lds_dst[2];
#pragma unroll
  for (int q = 0; q < 2; ++q)
    lds_dst[q] = (unsigned)(size_t)(const void*)&sA[a_r * A_STRIDE + 4 * (a_q0 + q)];

  for (int k0 = 0; k0 < FF; k0 += TILE_K) {
    // ---- stage A: bf16 act tile, async copy straight into LDS ----
#pragma unroll
    for (int q = 0; q < 2; ++q) {
      const unsigned long long ga =
          (unsigned long long)(const void*)(act + (size_t)(m0 + a_r) * FF + k0 + 8 * (a_q0 + q));
      asm volatile("global_load_async_to_lds_b128 %0, %1, off"
                   :: "v"(lds_dst[q]), "v"(ga) : "memory");
    }
    // ---- stage B, f32 -> bf16, transposed [n][K] ----
    {
      const float* p = W + (size_t)(k0 + b_kb) * HID + c0 + b_n;
      if (k0 + TILE_K < FF)
        __builtin_prefetch(p + (size_t)TILE_K * HID, 0, 1);
      float b[8];
#pragma unroll
      for (int i = 0; i < 8; ++i) b[i] = p[(size_t)i * HID];
#pragma unroll
      for (int i = 0; i < 4; ++i)
        sB[b_n * B_STRIDE + (b_kb >> 1) + i] = pack2(b[2 * i], b[2 * i + 1]);
    }
    // our async LDS writes must land before the barrier
    asm volatile("s_wait_asynccnt 0" ::: "memory");
    __syncthreads();

    v16bf af[2], bf[2];
#pragma unroll
    for (int i = 0; i < 2; ++i) af[i] = frag_a(sA, wm * 32 + i * 16, lane);
#pragma unroll
    for (int j = 0; j < 2; ++j) bf[j] = frag_b(sB, wn * 32 + j * 16, lane);
#pragma unroll
    for (int i = 0; i < 2; ++i)
#pragma unroll
      for (int j = 0; j < 2; ++j)
        acc[i][j] = wmma_bf16(af[i], bf[j], acc[i][j]);
    __syncthreads();
  }

  const int rhi = (lane >> 4) * 8;
  const int nc  = lane & 15;
#pragma unroll
  for (int i = 0; i < 2; ++i)
#pragma unroll
    for (int j = 0; j < 2; ++j) {
      const int gm = m0 + wm * 32 + i * 16 + rhi;
      const int gn = c0 + wn * 32 + j * 16 + nc;
#pragma unroll
      for (int v = 0; v < 8; ++v)
        out[(size_t)(gm + v) * HID + gn] = acc[i][j][v];
    }
}

// ---------------------------------------------------------------------------
extern "C" void kernel_launch(void* const* d_in, const int* in_sizes, int n_in,
                              void* d_out, int out_size, void* d_ws, size_t ws_size,
                              hipStream_t stream) {
  (void)in_sizes; (void)n_in; (void)out_size; (void)ws_size;
  const float* x   = (const float*)d_in[0];          // [N, HID]
  const int*   tpe = (const int*)d_in[1];            // [E]
  const float* w1  = (const float*)d_in[2];          // [E, HID, 2*FF]
  const float* w2  = (const float*)d_in[3];          // [E, FF, HID]
  float* out = (float*)d_out;                        // [N, HID]
  unsigned short* act = (unsigned short*)d_ws;       // [N, FF] bf16, 128 MB

  dim3 blk(256, 1, 1);
  dim3 g1(NTOK / TILE_M, FF / TILE_N, 1);    // 128 x 64
  dim3 g2(NTOK / TILE_M, HID / TILE_N, 1);   // 128 x 32

  moe_fc1_swiglu<<<g1, blk, 0, stream>>>(x, tpe, w1, act);
  moe_fc2<<<g2, blk, 0, stream>>>(act, tpe, w2, out);
}
